// TinyRecursiveReasoningModel_ACTV1Block_69612829933646
// MI455X (gfx1250) — matrix operand order repaired and checked by reading, not compiled, not run on hardware
//
#include <hip/hip_runtime.h>
#include <hip/hip_bf16.h>

typedef __attribute__((ext_vector_type(16))) __bf16 v16bf;
typedef __attribute__((ext_vector_type(8)))  __bf16 v8bf;
typedef __attribute__((ext_vector_type(8)))  float  v8f;
typedef __attribute__((ext_vector_type(4)))  unsigned int v4u;
typedef __attribute__((ext_vector_type(8)))  int v8i;
typedef __attribute__((ext_vector_type(4)))  int v4i;

__device__ __forceinline__ float silu_f(float x)     { return x / (1.f + __expf(-x)); }
__device__ __forceinline__ float softplus_f(float x) { return (x > 20.f) ? x : log1pf(__expf(x)); }

// ---------------- Tensor Data Mover: 2D tile (rows x cols bf16) global -> LDS ----------------
// D# per CDNA5 ISA: group0 = {count, lds_addr, global_addr, type=2},
// group1 = {data_size=2B, tensor_dim0=row_stride, tensor_dim1=rows, tile_dim0=cols,
//           tile_dim1=rows, tensor_dim0_stride=row_stride}. Groups 2/3 zero (dims unused).
__device__ __forceinline__ void tdm_load_2d_bf16(unsigned lds_byte_off, const void* gptr,
                                                 unsigned tile_cols, unsigned tile_rows,
                                                 unsigned row_stride /*elements*/) {
  unsigned long long ga = (unsigned long long)(size_t)gptr;
  v4u g0;
  g0[0] = 1u;                                                  // count=1, user mode
  g0[1] = lds_byte_off;                                        // lds_addr (bytes)
  g0[2] = (unsigned)ga;                                        // global_addr[31:0]
  g0[3] = (unsigned)((ga >> 32) & 0x01FFFFFFu) | (2u << 30);   // global_addr[56:32] | type=2
  v8i g1;
  g1[0] = (int)(1u << 16);                                     // data_size=1 -> 2 bytes
  g1[1] = (int)((row_stride & 0xFFFFu) << 16);                 // tensor_dim0[15:0]
  g1[2] = (int)(((row_stride >> 16) & 0xFFFFu) |
                ((tile_rows & 0xFFFFu) << 16));                // tensor_dim0[31:16] | tensor_dim1[15:0]
  g1[3] = (int)((tile_cols & 0xFFFFu) << 16);                  // tensor_dim1[31:16]=0 | tile_dim0
  g1[4] = (int)(tile_rows & 0xFFFFu);                          // tile_dim1 | tile_dim2=0
  g1[5] = (int)row_stride;                                     // tensor_dim0_stride[31:0]
  g1[6] = 0;                                                   // stride0[47:32] | stride1[15:0]
  g1[7] = 0;
  v4i z4 = {0, 0, 0, 0};
#if defined(__clang_major__) && (__clang_major__ >= 23)
  v8i z8 = {0, 0, 0, 0, 0, 0, 0, 0};
  __builtin_amdgcn_tensor_load_to_lds(g0, g1, z4, z4, z8, 0);
#else
  __builtin_amdgcn_tensor_load_to_lds(g0, g1, z4, z4, 0);
#endif
}

// ---------------- WMMA fragment loaders (bf16, 16x16x32) ----------------
// A fragment from LDS tile (16 rows x K, row-major):
// lanes 0-15: M=lane, K=0..15; lanes 16-31: same M, K=16..31.
__device__ __forceinline__ v16bf load_a_frag_lds(const __bf16* __restrict__ S, int K, int k0) {
  int lane = threadIdx.x & 31;
  int half = lane >> 4;
  const __bf16* row = S + (size_t)(lane & 15) * K;
  v16bf f;
  if (k0 + 32 <= K) {
    v8bf lo = *(const v8bf*)(row + k0 + half * 16);
    v8bf hi = *(const v8bf*)(row + k0 + half * 16 + 8);
#pragma unroll
    for (int e = 0; e < 8; ++e) { f[e] = lo[e]; f[8 + e] = hi[e]; }
  } else {
#pragma unroll
    for (int e = 0; e < 16; ++e) {
      int k = k0 + half * 16 + e;
      f[e] = (k < K) ? row[k] : (__bf16)0.f;
    }
  }
  return f;
}

// B: 32x16 (KxN) from W (N x K row-major): lane holds col=lane&15, K contiguous (+16 for hi half).
__device__ __forceinline__ v16bf load_b_frag(const __bf16* __restrict__ W, int ldw,
                                             int n0, int k0, int K) {
  int lane = threadIdx.x & 31;
  int half = lane >> 4;
  const __bf16* row = W + (size_t)(n0 + (lane & 15)) * ldw;
  v16bf f;
  if (k0 + 32 <= K) {
    f = *(const v16bf*)(row + k0 + half * 16);
  } else {
#pragma unroll
    for (int e = 0; e < 16; ++e) {
      int k = k0 + half * 16 + e;
      f[e] = (k < K) ? row[k] : (__bf16)0.f;
    }
  }
  return f;
}

// ---------------- WMMA GEMM:  C[M,N] = act(A[M,K] * W[N,K]^T + bias) ----------------
// Block: 4 waves; wave tile = 16 x (NT*16) with the A fragment reused across NT WMMAs.
// A tile (16 x K) staged to LDS by the Tensor Data Mover (wave 0), published via barrier.
// NT=4 requires N % 256 == 0 (guard-free inner loop, EXEC all-ones for WMMA).
// NT=1 handles ragged N via a single readfirstlane scalar guard outside the K loop.
// act: 0=none, 1=silu, 2=softplus
template <int NT>
__global__ __launch_bounds__(128) void wmma_gemm_kernel(
    const __bf16* __restrict__ A, int lda,
    const __bf16* __restrict__ W, int ldw,
    const float* __restrict__ bias,
    float* __restrict__ C, int ldc,
    int M, int N, int K, int act)
{
  extern __shared__ __bf16 sA[];          // 16 x K bf16 tile
  int m0 = blockIdx.y * 16;

  // Wave 0 issues the TDM transfer (EXEC-independent, per-wave op), waits TENSORcnt,
  // then the workgroup barrier publishes the LDS tile to all waves.
  if (__builtin_amdgcn_readfirstlane(threadIdx.y) == 0) {
    tdm_load_2d_bf16(__builtin_amdgcn_groupstaticsize(),
                     A + (size_t)m0 * lda, (unsigned)K, 16u, (unsigned)lda);
    __builtin_amdgcn_s_wait_tensorcnt(0);
  }
  __syncthreads();

  int nbase = (blockIdx.x * 4 + threadIdx.y) * (NT * 16);
  // Scalar (SGPR) guard -> s_cbranch, no EXEC masking; placed after the only barrier.
  if (__builtin_amdgcn_readfirstlane((nbase < N) ? 1 : 0) == 0) return;

  v8f acc[NT];
#pragma unroll
  for (int j = 0; j < NT; ++j) acc[j] = (v8f){0.f, 0.f, 0.f, 0.f, 0.f, 0.f, 0.f, 0.f};

  int k0 = 0;
  for (; k0 + 32 <= K; k0 += 32) {
    v16bf a = load_a_frag_lds(sA, K, k0);
#pragma unroll
    for (int j = 0; j < NT; ++j) {
      v16bf b = load_b_frag(W, ldw, nbase + j * 16, k0, K);
      acc[j] = __builtin_amdgcn_wmma_f32_16x16x32_bf16(false, a, false, b, (short)0,
                                                       acc[j], false, false);
    }
  }
  if (k0 < K) {                           // K tail (K=48 case)
    v16bf a = load_a_frag_lds(sA, K, k0);
#pragma unroll
    for (int j = 0; j < NT; ++j) {
      v16bf b = load_b_frag(W, ldw, nbase + j * 16, k0, K);
      acc[j] = __builtin_amdgcn_wmma_f32_16x16x32_bf16(false, a, false, b, (short)0,
                                                       acc[j], false, false);
    }
  }

  // C/D layout: VGPR r -> M = r + 8*(lane>=16), N = lane&15
  int lane = threadIdx.x & 31;
  int mb   = m0 + ((lane >> 4) << 3);
#pragma unroll
  for (int j = 0; j < NT; ++j) {
    int col = nbase + j * 16 + (lane & 15);
    float bv = bias ? bias[col] : 0.f;
#pragma unroll
    for (int r = 0; r < 8; ++r) {
      float v = acc[j][r] + bv;
      if (act == 1)      v = silu_f(v);
      else if (act == 2) v = softplus_f(v);
      C[(size_t)(mb + r) * ldc + col] = v;
    }
  }
}

// ---------------- elementwise helpers ----------------
__global__ void cvt_f32_bf16_kernel(const float* __restrict__ src, __bf16* __restrict__ dst, long n) {
  long i = (long)blockIdx.x * blockDim.x + threadIdx.x;
  long stride = (long)gridDim.x * blockDim.x;
  for (; i < n; i += stride) dst[i] = (__bf16)src[i];
}

__global__ void cvt_slice_kernel(const float* __restrict__ src, int src_ld,
                                 __bf16* __restrict__ dst, int dst_ld,
                                 long rows, int cols) {
  long total = rows * cols;
  long i = (long)blockIdx.x * blockDim.x + threadIdx.x;
  if (i >= total) return;
  long r = i / cols; int c = (int)(i % cols);
  dst[r * dst_ld + c] = (__bf16)src[r * src_ld + c];
}

// depthwise causal (dir=0) / anti-causal (dir=1) conv over xs = xz[:, :DI], + bias + silu
__global__ void dwconv_silu_kernel(const float* __restrict__ xz, const float* __restrict__ cw,
                                   const float* __restrict__ cb, float* __restrict__ u,
                                   __bf16* __restrict__ ub, int Bn, int L, int DI2, int DI, int dir)
{
  long total = (long)Bn * L * DI;
  long i = (long)blockIdx.x * blockDim.x + threadIdx.x;
  if (i >= total) return;
  int c = (int)(i % DI);
  long bt = i / DI;
  int t = (int)(bt % L);
  int b = (int)(bt / L);
  float acc = cb[c];
#pragma unroll
  for (int j = 0; j < 4; ++j) {
    int tt = dir ? (t + 3 - j) : (t - 3 + j);
    if (tt >= 0 && tt < L)
      acc += cw[c * 4 + j] * xz[(size_t)(b * L + tt) * DI2 + c];
  }
  float v = silu_f(acc);
  u[i] = v;
  ub[i] = (__bf16)v;
}

// selective scan: one thread per (batch, channel); h[16] in registers.
// dir=1 walks t = L-1..0 (flip), writing aligned original positions; accum adds fwd+bwd.
__global__ void scan_kernel(const float* __restrict__ u, const float* __restrict__ delta,
                            const float* __restrict__ dbl, int ld_dbl, int offB, int offC,
                            const float* __restrict__ A_log, const float* __restrict__ Dsk,
                            float* __restrict__ y, int Bn, int L, int DI, int dir, int accum)
{
  int i = blockIdx.x * blockDim.x + threadIdx.x;
  if (i >= Bn * DI) return;
  int d = i % DI;
  int b = i / DI;
  float Av[16], h[16];
#pragma unroll
  for (int nn = 0; nn < 16; ++nn) { Av[nn] = -__expf(A_log[d * 16 + nn]); h[nn] = 0.f; }
  float Dv = Dsk[d];
  for (int s = 0; s < L; ++s) {
    int t = dir ? (L - 1 - s) : s;
    size_t row = (size_t)b * L + t;
    float ut = u[row * DI + d];
    float dt = delta[row * DI + d];
    const float* Bv = dbl + row * ld_dbl + offB;
    const float* Cv = dbl + row * ld_dbl + offC;
    float yv = 0.f;
#pragma unroll
    for (int nn = 0; nn < 16; ++nn) {
      float dA = __expf(dt * Av[nn]);
      h[nn] = h[nn] * dA + dt * Bv[nn] * ut;
      yv += h[nn] * Cv[nn];
    }
    float o = yv + ut * Dv;
    if (accum) y[row * DI + d] += o; else y[row * DI + d] = o;
  }
}

// yg = bf16( 0.5 * (y_fwd + y_bwd) * silu(z) ),  z = xz[:, DI:]
__global__ void gate_kernel(const float* __restrict__ y, const float* __restrict__ xz,
                            __bf16* __restrict__ yg, long total, int DI, int DI2) {
  long i = (long)blockIdx.x * blockDim.x + threadIdx.x;
  if (i >= total) return;
  int c = (int)(i % DI);
  long row = i / DI;
  float z = xz[row * DI2 + DI + c];
  yg[i] = (__bf16)(0.5f * y[i] * silu_f(z));
}

// out = rmsnorm(a + b); optional f32 and bf16 outputs. One block (256 thr) per row.
__global__ __launch_bounds__(256) void add_rmsnorm_kernel(const float* __restrict__ a,
                                                          const float* __restrict__ b,
                                                          float* __restrict__ of,
                                                          __bf16* __restrict__ ob, int width)
{
  int row = blockIdx.x;
  __shared__ float sred[256];
  const float* ar = a + (size_t)row * width;
  const float* br = b + (size_t)row * width;
  float local = 0.f;
  for (int i = threadIdx.x; i < width; i += 256) { float v = ar[i] + br[i]; local += v * v; }
  sred[threadIdx.x] = local;
  __syncthreads();
  for (int sft = 128; sft > 0; sft >>= 1) {
    if (threadIdx.x < sft) sred[threadIdx.x] += sred[threadIdx.x + sft];
    __syncthreads();
  }
  float scale = rsqrtf(sred[0] / width + 1e-5f);
  for (int i = threadIdx.x; i < width; i += 256) {
    float v = (ar[i] + br[i]) * scale;
    if (of) of[(size_t)row * width + i] = v;
    if (ob) ob[(size_t)row * width + i] = (__bf16)v;
  }
}

// sb = bf16( silu(gu[:, :inter]) * gu[:, inter:] )
__global__ void swiglu_kernel(const float* __restrict__ gu, __bf16* __restrict__ sb,
                              long total, int inter) {
  long i = (long)blockIdx.x * blockDim.x + threadIdx.x;
  if (i >= total) return;
  int c = (int)(i % inter);
  long row = i / inter;
  const float* g = gu + row * (size_t)(2 * inter);
  sb[i] = (__bf16)(silu_f(g[c]) * g[inter + c]);
}

// ---------------- host orchestration ----------------
extern "C" void kernel_launch(void* const* d_in, const int* in_sizes, int n_in,
                              void* d_out, int out_size, void* d_ws, size_t ws_size,
                              hipStream_t stream)
{
  const float* x       = (const float*)d_in[0];
  const float* in_w    = (const float*)d_in[1];
  const float* conv_w  = (const float*)d_in[2];
  const float* conv_b  = (const float*)d_in[3];
  const float* xproj_w = (const float*)d_in[4];
  const float* dt_w    = (const float*)d_in[5];
  const float* dt_b    = (const float*)d_in[6];
  const float* A_log   = (const float*)d_in[7];
  const float* Dsk     = (const float*)d_in[8];
  const float* out_w   = (const float*)d_in[9];
  const float* guw     = (const float*)d_in[10];
  const float* down_w  = (const float*)d_in[11];
  float* out = (float*)d_out;

  const int Bn = 4, L = 1024, Dm = 768, BL = 4096;
  const int DI = 1536, DI2 = 3072, DTR = 48, DBLD = 80, IN = 2048;

  char* base = (char*)d_ws;
  size_t o = 0;
  auto take = [&](size_t nbytes) -> void* {
    void* p = base + o;
    o = (o + nbytes + 255) & ~(size_t)255;
    return p;
  };

  // bf16 copies of activations/weights
  __bf16* xb     = (__bf16*)take((size_t)BL * Dm * 2);
  __bf16* wb_in  = (__bf16*)take((size_t)DI2 * Dm * 2);
  __bf16* wb_xp  = (__bf16*)take((size_t)DBLD * DI * 2);
  __bf16* wb_dt  = (__bf16*)take((size_t)DI * DTR * 2);
  __bf16* wb_out = (__bf16*)take((size_t)Dm * DI * 2);
  __bf16* wb_gu  = (__bf16*)take((size_t)2 * IN * Dm * 2);
  __bf16* wb_dn  = (__bf16*)take((size_t)Dm * IN * 2);

  // arena1: xz (50.3MB, live until gating) then gu (67MB, after gating)
  char* a1 = (char*)take(67108864);
  float* xz = (float*)a1;
  float* gu = (float*)a1;
  // arena2: u (25.2MB, per-direction) then h (12.6MB f32) + hb (6.3MB bf16)
  char* a2 = (char*)take(25165824);
  float* u  = (float*)a2;
  float* h  = (float*)a2;
  __bf16* hb = (__bf16*)(a2 + 12582912);
  // arena3: ub (bf16 12.6MB) then ymamba (f32) then mlpout (f32)
  char* a3 = (char*)take(12582912);
  __bf16* ub  = (__bf16*)a3;
  float* ymm  = (float*)a3;
  float* mlpo = (float*)a3;
  float*  dbl = (float*)take((size_t)BL * DBLD * 4);
  __bf16* dtb = (__bf16*)take((size_t)BL * DTR * 2);
  // arena5: delta (25.2MB, per-direction) then yg (bf16)
  char* a5 = (char*)take(25165824);
  float* delta = (float*)a5;
  __bf16* yg   = (__bf16*)a5;
  // arena6: y accumulator (25.2MB) then sb (bf16 16.8MB)
  char* a6 = (char*)take(25165824);
  float* ybuf = (float*)a6;
  __bf16* sb  = (__bf16*)a6;

  auto cvt = [&](const float* s, __bf16* d, long n) {
    cvt_f32_bf16_kernel<<<dim3(2048), dim3(256), 0, stream>>>(s, d, n);
  };
  auto gemm = [&](const __bf16* A, int lda, const __bf16* W, int ldw, const float* bias,
                  float* C, int ldc, int M, int N, int K, int act) {
    dim3 blk(32, 4);
    size_t shmem = (size_t)K * 16 * 2;   // 16 x K bf16 A tile for the TDM
    if (N % 256 == 0) {
      dim3 grd(N / 256, M / 16);
      wmma_gemm_kernel<4><<<grd, blk, shmem, stream>>>(A, lda, W, ldw, bias, C, ldc, M, N, K, act);
    } else {
      dim3 grd((N + 63) / 64, M / 16);
      wmma_gemm_kernel<1><<<grd, blk, shmem, stream>>>(A, lda, W, ldw, bias, C, ldc, M, N, K, act);
    }
  };

  // 1. precision conversion (weights + x)
  cvt(x, xb, (long)BL * Dm);
  cvt(in_w, wb_in, (long)DI2 * Dm);
  cvt(xproj_w, wb_xp, (long)DBLD * DI);
  cvt(dt_w, wb_dt, (long)DI * DTR);
  cvt(out_w, wb_out, (long)Dm * DI);
  cvt(guw, wb_gu, (long)2 * IN * Dm);
  cvt(down_w, wb_dn, (long)Dm * IN);

  // 2. xz = x @ in_proj^T  (shared by both directions — in_proj is position-independent)
  gemm(xb, Dm, wb_in, Dm, nullptr, xz, DI2, BL, DI2, Dm, 0);

  // 3. per-direction mamba inner pipeline; scan accumulates fwd+bwd into ybuf
  for (int dir = 0; dir < 2; ++dir) {
    long tc = (long)BL * DI;
    dwconv_silu_kernel<<<(tc + 255) / 256, 256, 0, stream>>>(xz, conv_w, conv_b, u, ub,
                                                             Bn, L, DI2, DI, dir);
    gemm(ub, DI, wb_xp, DI, nullptr, dbl, DBLD, BL, DBLD, DI, 0);
    long ts = (long)BL * DTR;
    cvt_slice_kernel<<<(ts + 255) / 256, 256, 0, stream>>>(dbl, DBLD, dtb, DTR, BL, DTR);
    gemm(dtb, DTR, wb_dt, DTR, dt_b, delta, DI, BL, DI, DTR, 2);   // fused +bias, softplus
    int sc = Bn * DI;
    scan_kernel<<<(sc + 255) / 256, 256, 0, stream>>>(u, delta, dbl, DBLD, DTR, DTR + 16,
                                                      A_log, Dsk, ybuf, Bn, L, DI, dir, dir);
  }

  // 4. gate with silu(z), single out_proj GEMM for both directions
  long tg = (long)BL * DI;
  gate_kernel<<<(tg + 255) / 256, 256, 0, stream>>>(ybuf, xz, yg, tg, DI, DI2);
  gemm(yg, DI, wb_out, DI, nullptr, ymm, Dm, BL, Dm, DI, 0);

  // 5. h = rmsnorm(x + y_mamba)
  add_rmsnorm_kernel<<<BL, 256, 0, stream>>>(x, ymm, h, hb, Dm);

  // 6. SwiGLU MLP
  gemm(hb, Dm, wb_gu, Dm, nullptr, gu, 2 * IN, BL, 2 * IN, Dm, 0);
  long tw = (long)BL * IN;
  swiglu_kernel<<<(tw + 255) / 256, 256, 0, stream>>>(gu, sb, tw, IN);
  gemm(sb, IN, wb_dn, IN, nullptr, mlpo, Dm, BL, Dm, IN, 0);

  // 7. out = rmsnorm(h + mlp_out)
  add_rmsnorm_kernel<<<BL, 256, 0, stream>>>(h, mlpo, out, nullptr, Dm);

  (void)in_sizes; (void)n_in; (void)out_size; (void)ws_size;
}
